// SwinUNet_51479478010544
// MI455X (gfx1250) — matrix operand order repaired and checked by reading, not compiled, not run on hardware
//
#include <hip/hip_runtime.h>
#include <hip/hip_bf16.h>
#include <math.h>

// ---------------------------------------------------------------------------
// Swin block ×2 (adaLN-modulated, shifted windows) for MI455X / gfx1250.
// fp16 WMMA (v_wmma_f32_16x16x32_f16), f32 accumulate.
// Memory-bound (~112 GFLOP, ~400MB @ 23.3 TB/s): all operand fetches are
// wide contiguous loads — weights pre-packed in B-fragment order (32B/lane),
// V pre-transposed for P@V, A staged via LDS (ds_load_b128 fragments).
// Workspace requirement: ~88 MB.
// ---------------------------------------------------------------------------

typedef __attribute__((ext_vector_type(16))) _Float16 v16h;
typedef __attribute__((ext_vector_type(8)))  _Float16 v8h;
typedef __attribute__((ext_vector_type(4)))  _Float16 v4h;
typedef __attribute__((ext_vector_type(8)))  float    v8f;

#define LANE (threadIdx.x & 31)
#define WAVE (threadIdx.x >> 5)

static constexpr int M_TOTAL = 32768;   // 8 * 64 * 64 tokens
static constexpr int DIMC    = 256;
static constexpr float ATT_SCALE = 0.17677669529663687f; // 32^-0.5

__device__ __forceinline__ v8f wmma16(v16h a, v16h b, v8f c) {
  return __builtin_amdgcn_wmma_f32_16x16x32_f16(false, a, false, b, (short)0, c,
                                                false, false);
}
__device__ __forceinline__ v16h cat8(v8h lo, v8h hi) {
  return __builtin_shufflevector(lo, hi, 0,1,2,3,4,5,6,7,8,9,10,11,12,13,14,15);
}

__device__ __forceinline__ float wave_sum32(float v) {
#pragma unroll
  for (int m = 16; m > 0; m >>= 1) v += __shfl_xor(v, m, 32);
  return v;
}
__device__ __forceinline__ float half_max16(float v) {
#pragma unroll
  for (int m = 8; m > 0; m >>= 1) v = fmaxf(v, __shfl_xor(v, m, 32));
  return v;
}
__device__ __forceinline__ float half_sum16(float v) {
#pragma unroll
  for (int m = 8; m > 0; m >>= 1) v += __shfl_xor(v, m, 32);
  return v;
}

// window-order row m -> original row (b*4096 + y*64 + x), applying roll by +shift
__device__ __forceinline__ int win_to_orig(int m, int shift) {
  int wi = m >> 6, t = m & 63;
  int b = wi >> 6, wy = (wi >> 3) & 7, wx = wi & 7;
  int y = ((wy << 3) + (t >> 3) + shift) & 63;
  int x = ((wx << 3) + (t & 7) + shift) & 63;
  return (b << 12) + (y << 6) + x;
}

// ---------------- small prep kernels ----------------
__global__ void k_silu(const float* __restrict__ emb, float* __restrict__ out) {
  int i = blockIdx.x * 256 + threadIdx.x;
  if (i < 8 * 512) { float v = emb[i]; out[i] = v / (1.f + expf(-v)); }
}

// ada[blk][b][1536] = silu(emb[b]) @ ada_w[blk] + ada_b[blk]
__global__ void k_ada(const float* __restrict__ silu, const float* __restrict__ aw,
                      const float* __restrict__ ab, float* __restrict__ ada) {
  int idx = blockIdx.x * 256 + threadIdx.x;      // 2*8*1536 = 24576 threads
  int blk = idx / 12288, rem = idx % 12288;
  int b = rem / 1536, j = rem % 1536;
  float acc = ab[blk * 1536 + j];
  const float* s = silu + b * 512;
  const float* w = aw + (size_t)blk * 512 * 1536 + j;
  for (int e = 0; e < 512; e++) acc += s[e] * w[(size_t)e * 1536];
  ada[(size_t)(blk * 8 + b) * 1536 + j] = acc;
}

// f32 W[K][N] -> f16 packed in WMMA-B-fragment order: Wp[k>>4][n][k&15]
// so a lane's full 16-half B fragment is one contiguous 32-byte run.
__global__ void k_cvt_pack(const float* __restrict__ s, _Float16* __restrict__ d,
                           int N, int n_total) {
  int i = blockIdx.x * 256 + threadIdx.x;
  if (i < n_total) {
    int k = i / N, n = i - k * N;
    d[(((size_t)(k >> 4) * N) + n) * 16 + (k & 15)] = (_Float16)s[i];
  }
}

// ---------------- LN + adaLN modulate + (optional) window gather ----------------
__global__ void k_modln(const float* __restrict__ X, _Float16* __restrict__ outA,
                        const float* __restrict__ ada, int blk, int shift,
                        int do_window, int ch_sh, int ch_sc) {
  int m = blockIdx.x * 8 + WAVE;                 // destination row
  int src = do_window ? win_to_orig(m, shift) : m;
  int b = src >> 12;
  const float* xp = X + (size_t)src * DIMC;
  float xv[8], s = 0.f;
#pragma unroll
  for (int j = 0; j < 8; j++) { xv[j] = xp[LANE + 32 * j]; s += xv[j]; }
  float mu = wave_sum32(s) * (1.f / 256.f);
  float vr = 0.f;
#pragma unroll
  for (int j = 0; j < 8; j++) { float d = xv[j] - mu; vr += d * d; }
  float rstd = rsqrtf(wave_sum32(vr) * (1.f / 256.f) + 1e-6f);
  const float* sh = ada + ((size_t)(blk * 8 + b) * 6 + ch_sh) * 256;
  const float* sc = ada + ((size_t)(blk * 8 + b) * 6 + ch_sc) * 256;
#pragma unroll
  for (int j = 0; j < 8; j++) {
    int c = LANE + 32 * j;
    float o = (xv[j] - mu) * rstd * (1.f + sc[c]) + sh[c];
    outA[(size_t)m * DIMC + c] = (_Float16)o;
  }
}

// ---------------- generic WMMA GEMM, LDS-staged A, packed B, fused epilogues ----
// mode 0: C16 = A@B + bias (qkv); V columns (n>=512) stored transposed to Vt
// mode 1: X[map(m)] += g_sa * (A@B + bias)    (proj + roll-back + residual)
// mode 2: C16 = gelu(A@B + bias)              (ff1)
// mode 3: X[m]      += g_ff * (A@B + bias)    (ff2 + residual)
__global__ __launch_bounds__(256) void k_gemm(
    const _Float16* __restrict__ A, const _Float16* __restrict__ Bp,
    const float* __restrict__ bias, _Float16* __restrict__ Cout,
    _Float16* __restrict__ Vt, float* __restrict__ X,
    const float* __restrict__ ada,
    int N, int K, int mode, int blk, int shift) {
  __shared__ _Float16 sA[32][40];                // 32 rows x 32 k, padded
  int ln = LANE & 15, hf = LANE >> 4;
  int wm = WAVE & 1, wn = WAVE >> 1;
  int mblk = blockIdx.y * 32;
  int m0 = mblk + wm * 16;
  int n0 = blockIdx.x * 64 + wn * 16;
  int n = n0 + ln;
  v8f acc = {};
  for (int kk = 0; kk < K; kk += 32) {
    {                                            // cooperative A stage, b64/thread
      int row = threadIdx.x >> 3, cp = (threadIdx.x & 7) << 2;
      *(v4h*)&sA[row][cp] = *(const v4h*)&A[(size_t)(mblk + row) * K + kk + cp];
    }
    if (kk + 32 < K)                             // gfx1250 global_prefetch path
      __builtin_prefetch(&A[(size_t)(mblk + (threadIdx.x & 31)) * K + kk + 32], 0, 3);
    __syncthreads();
    // A frag: lane=M; per-lane halves are two contiguous 16B runs in LDS
    v8h a0 = *(const v8h*)&sA[wm * 16 + ln][hf << 4];
    v8h a1 = *(const v8h*)&sA[wm * 16 + ln][(hf << 4) + 8];
    // B frag: one contiguous 32-byte run in packed weights
    v16h b = *(const v16h*)&Bp[(((size_t)(kk >> 4) + hf) * N + n) * 16];
    acc = wmma16(cat8(a0, a1), b, acc);
    __syncthreads();
  }
#pragma unroll
  for (int r = 0; r < 8; r++) {
    int m = m0 + (hf << 3) + r;
    float val = acc[r] + bias[n];
    if (mode == 0) {
      if (n < 512) {
        Cout[(size_t)m * N + n] = (_Float16)val;      // Q,K rows
      } else {                                        // V stored transposed
        Vt[((size_t)(m >> 6) * 256 + (n - 512)) * 64 + (m & 63)] = (_Float16)val;
      }
    } else if (mode == 2) {
      val = 0.5f * val * (1.f + erff(val * 0.70710678118f));
      Cout[(size_t)m * N + n] = (_Float16)val;
    } else if (mode == 1) {
      int src = win_to_orig(m, shift);
      float g = ada[((size_t)(blk * 8 + (m >> 12)) * 6 + 2) * 256 + n];
      X[(size_t)src * DIMC + n] += g * val;
    } else {
      float g = ada[((size_t)(blk * 8 + (m >> 12)) * 6 + 5) * 256 + n];
      X[(size_t)m * DIMC + n] += g * val;
    }
  }
}

// ---------------- fused window attention: one wave per (window, head) ----------------
__global__ __launch_bounds__(256) void k_attn(const _Float16* __restrict__ qkv,
                                              const _Float16* __restrict__ Vt,
                                              _Float16* __restrict__ outA,
                                              const float* __restrict__ btab,
                                              int shift) {
  __shared__ _Float16 sP[8][16 * 64];            // per-wave P staging (C-frag -> A-frag)
  int win = blockIdx.x;                          // 512 windows
  int h = WAVE;                                  // 8 heads
  int ln = LANE & 15, hf = LANE >> 4;
  int wy = (win >> 3) & 7, wx = win & 7;
  const _Float16* base = qkv + (size_t)win * 64 * 768;
  const _Float16* vbase = Vt + ((size_t)win * 256 + h * 32) * 64;

  v16h kb[4];                                    // K^T B-frags: contiguous 32B/lane
#pragma unroll
  for (int nj = 0; nj < 4; nj++)
    kb[nj] = *(const v16h*)(base + (size_t)(nj * 16 + ln) * 768 + 256 + h * 32 + (hf << 4));
  v16h vb[4];                                    // V B-frags from Vt: contiguous 32B/lane
#pragma unroll
  for (int kj = 0; kj < 2; kj++)
#pragma unroll
    for (int dj = 0; dj < 2; dj++)
      vb[kj * 2 + dj] =
          *(const v16h*)(vbase + (size_t)(dj * 16 + ln) * 64 + kj * 32 + (hf << 4));

  for (int mi = 0; mi < 4; mi++) {               // 16-query row blocks
    const _Float16* qp = base + (size_t)(mi * 16 + ln) * 768 + h * 32;
    v16h qa = cat8(*(const v8h*)(qp + (hf << 4)), *(const v8h*)(qp + (hf << 4) + 8));
    v8f st[4];
#pragma unroll
    for (int nj = 0; nj < 4; nj++) { v8f z = {}; st[nj] = wmma16(qa, kb[nj], z); }

    // scale + relative-position bias + shift mask (computed inline)
#pragma unroll
    for (int nj = 0; nj < 4; nj++)
#pragma unroll
      for (int r = 0; r < 8; r++) {
        int qi = mi * 16 + (hf << 3) + r;
        int ki = nj * 16 + ln;
        int qy = qi >> 3, qx = qi & 7, ky = ki >> 3, kx = ki & 7;
        int relidx = (qy - ky + 7) * 15 + (qx - kx + 7);
        float v = st[nj][r] * ATT_SCALE + btab[relidx * 8 + h];
        if (shift > 0) {
          int yq = wy * 8 + qy, xq = wx * 8 + qx;
          int yk = wy * 8 + ky, xk = wx * 8 + kx;
          int lq = (yq < 56 ? 0 : (yq < 60 ? 1 : 2)) * 3 + (xq < 56 ? 0 : (xq < 60 ? 1 : 2));
          int lk = (yk < 56 ? 0 : (yk < 60 ? 1 : 2)) * 3 + (xk < 56 ? 0 : (xk < 60 ? 1 : 2));
          if (lq != lk) v -= 100.f;
        }
        st[nj][r] = v;
      }

    // register softmax over 64 keys (rows live on 16-lane half groups)
#pragma unroll
    for (int r = 0; r < 8; r++) {
      float mx = fmaxf(fmaxf(st[0][r], st[1][r]), fmaxf(st[2][r], st[3][r]));
      mx = half_max16(mx);
      float sm = 0.f;
#pragma unroll
      for (int nj = 0; nj < 4; nj++) { float e = expf(st[nj][r] - mx); st[nj][r] = e; sm += e; }
      float inv = 1.f / half_sum16(sm);
      int rr = (hf << 3) + r;
#pragma unroll
      for (int nj = 0; nj < 4; nj++)
        sP[WAVE][rr * 64 + nj * 16 + ln] = (_Float16)(st[nj][r] * inv);
    }

    // O(mi) = P(16x64) @ V(64x32); P frags via ds_load_b128
    v16h pa[2];
#pragma unroll
    for (int kj = 0; kj < 2; kj++) {
      const _Float16* pp = &sP[WAVE][ln * 64 + kj * 32];
      pa[kj] = cat8(*(const v8h*)(pp + (hf << 4)), *(const v8h*)(pp + (hf << 4) + 8));
    }
#pragma unroll
    for (int dj = 0; dj < 2; dj++) {
      v8f o = {};
      o = wmma16(pa[0], vb[dj], o);
      o = wmma16(pa[1], vb[2 + dj], o);
#pragma unroll
      for (int r = 0; r < 8; r++) {
        int qi = mi * 16 + (hf << 3) + r;
        outA[(size_t)(win * 64 + qi) * DIMC + h * 32 + dj * 16 + ln] = (_Float16)o[r];
      }
    }
  }
}

// ---------------------------------------------------------------------------
extern "C" void kernel_launch(void* const* d_in, const int* in_sizes, int n_in,
                              void* d_out, int out_size, void* d_ws, size_t ws_size,
                              hipStream_t stream) {
  const float* x      = (const float*)d_in[0];
  const float* emb    = (const float*)d_in[1];
  const float* qkv_w  = (const float*)d_in[2];
  const float* qkv_b  = (const float*)d_in[3];
  const float* proj_w = (const float*)d_in[4];
  const float* proj_b = (const float*)d_in[5];
  const float* btab   = (const float*)d_in[6];
  const float* ff_w1  = (const float*)d_in[7];
  const float* ff_b1  = (const float*)d_in[8];
  const float* ff_w2  = (const float*)d_in[9];
  const float* ff_b2  = (const float*)d_in[10];
  const float* ada_w  = (const float*)d_in[11];
  const float* ada_b  = (const float*)d_in[12];

  float* X = (float*)d_out;                       // running activations (B,L,C) f32
  char* wsb = (char*)d_ws;
  float*    silu = (float*)(wsb);                 // 16 KB
  float*    ada  = (float*)(wsb + (16 << 10));    // 96 KB
  _Float16* W16  = (_Float16*)(wsb + (128 << 10));// 3 MB of packed fp16 weights
  _Float16* A16  = (_Float16*)(wsb + (4 << 20));  // 16 MB (h / attn_out / h2)
  _Float16* B16  = (_Float16*)(wsb + (21 << 20)); // 64 MB (qkv+Vt / ff1 out)
  _Float16* Vt   = B16 + (size_t)M_TOTAL * 768;   // 16 MB tail of B16 region
  _Float16* qkvW  = W16;
  _Float16* projW = W16 + 393216;
  _Float16* ff1W  = W16 + 524288;
  _Float16* ff2W  = W16 + 1048576;

  hipMemcpyAsync(X, x, (size_t)M_TOTAL * DIMC * sizeof(float),
                 hipMemcpyDeviceToDevice, stream);
  k_silu<<<16, 256, 0, stream>>>(emb, silu);
  k_ada<<<96, 256, 0, stream>>>(silu, ada_w, ada_b, ada);
  for (int i = 0; i < 2; i++) {                   // pack weights per block
    k_cvt_pack<<<768, 256, 0, stream>>>(qkv_w + (size_t)i * 196608,
                                        qkvW + (size_t)i * 196608, 768, 196608);
    k_cvt_pack<<<256, 256, 0, stream>>>(proj_w + (size_t)i * 65536,
                                        projW + (size_t)i * 65536, 256, 65536);
    k_cvt_pack<<<1024, 256, 0, stream>>>(ff_w1 + (size_t)i * 262144,
                                         ff1W + (size_t)i * 262144, 1024, 262144);
    k_cvt_pack<<<1024, 256, 0, stream>>>(ff_w2 + (size_t)i * 262144,
                                         ff2W + (size_t)i * 262144, 256, 262144);
  }

  for (int i = 0; i < 2; i++) {
    int shift = i ? 4 : 0;
    // h = LN(x)*(1+sc_sa)+sh_sa, rolled + window-partitioned, fp16
    k_modln<<<4096, 256, 0, stream>>>(X, A16, ada, i, shift, 1, 0, 1);
    // qkv = h @ qkv_w + qkv_b -> B16 (Q,K rows) + Vt (V transposed)
    k_gemm<<<dim3(12, 1024), 256, 0, stream>>>(A16, qkvW + (size_t)i * 196608,
        qkv_b + i * 768, B16, Vt, nullptr, nullptr, 768, 256, 0, i, 0);
    // fused window attention -> A16 (32768 x 256 f16)
    k_attn<<<512, 256, 0, stream>>>(B16, Vt, A16, btab + i * 1800, shift);
    // x += g_sa * (attn_out @ proj_w + proj_b), rolled back
    k_gemm<<<dim3(4, 1024), 256, 0, stream>>>(A16, projW + (size_t)i * 65536,
        proj_b + i * 256, nullptr, nullptr, X, ada, 256, 256, 1, i, shift);
    // h2 = LN(x)*(1+sc_ff)+sh_ff
    k_modln<<<4096, 256, 0, stream>>>(X, A16, ada, i, 0, 0, 3, 4);
    // ff1 = gelu(h2 @ w1 + b1) -> B16 (32768 x 1024 f16)
    k_gemm<<<dim3(16, 1024), 256, 0, stream>>>(A16, ff1W + (size_t)i * 262144,
        ff_b1 + i * 1024, B16, nullptr, nullptr, nullptr, 1024, 256, 2, i, 0);
    // x += g_ff * (ff1 @ w2 + b2)
    k_gemm<<<dim3(4, 1024), 256, 0, stream>>>(B16, ff2W + (size_t)i * 262144,
        ff_b2 + i * 256, nullptr, nullptr, X, ada, 256, 1024, 3, i, 0);
  }
}
